// K_Sparse_Autoencoder_30305289241005
// MI455X (gfx1250) — compile-verified, matrix operand order checked by reading, not run on hardware
//
#include <hip/hip_runtime.h>

#define INPUT_DIM  4096
#define BOTTLENECK 32768
#define BATCH      4096
#define K_COUNT    3276

typedef __bf16 bfloat;
typedef __attribute__((ext_vector_type(16))) __bf16 v16bf;
typedef __attribute__((ext_vector_type(8)))  __bf16 bf16x8;
typedef __attribute__((ext_vector_type(8)))  float  v8f;
typedef __attribute__((ext_vector_type(8)))  float  f32x8;

// ---------------------------------------------------------------------------
// Fragment builders.
// 16-bit A/B VGPR layout (ISA 7.12.2): lanes 0-15 hold K chunks [0..7] and
// [16..23]; lanes 16-31 hold [8..15] and [24..31]. "half" = lane>>4 selects
// which 8-wide chunk this lane loads at k0 and k0+16.
// ---------------------------------------------------------------------------
__device__ __forceinline__ v16bf frag_from_f32(const float* __restrict__ row,
                                               int k0, int half,
                                               float th, bool use_mask) {
  const f32x8 v0 = *(const f32x8*)(row + k0 + half * 8);
  const f32x8 v1 = *(const f32x8*)(row + k0 + 16 + half * 8);
  v16bf f;
#pragma unroll
  for (int i = 0; i < 8; i++) {
    float a = v0[i];
    float b = v1[i];
    if (use_mask) {
      a = (a >= th) ? a : 0.0f;
      b = (b >= th) ? b : 0.0f;
    }
    f[i]     = (__bf16)a;
    f[i + 8] = (__bf16)b;
  }
  return f;
}

__device__ __forceinline__ v16bf frag_from_bf16(const bfloat* __restrict__ row,
                                                int k0, int half) {
  const bf16x8 v0 = *(const bf16x8*)(row + k0 + half * 8);
  const bf16x8 v1 = *(const bf16x8*)(row + k0 + 16 + half * 8);
  v16bf f;
#pragma unroll
  for (int i = 0; i < 8; i++) {
    f[i]     = v0[i];
    f[i + 8] = v1[i];
  }
  return f;
}

// ---------------------------------------------------------------------------
// Kernel 1: WT[i][h] = bf16(W[h][i])   (32x32 tiles through LDS)
// ---------------------------------------------------------------------------
__global__ __launch_bounds__(256) void k_transpose(const float* __restrict__ W,
                                                   bfloat* __restrict__ WT) {
  __shared__ bfloat tile[32][36];
  const int tid = threadIdx.x;
  const int h0  = blockIdx.x * 32;
  const int i0  = blockIdx.y * 32;
  const int r   = tid >> 3;         // 0..31
  const int c   = (tid & 7) * 4;    // 0,4,...,28

  const float4 v = *(const float4*)(W + (size_t)(h0 + r) * INPUT_DIM + i0 + c);
  tile[r][c + 0] = (__bf16)v.x;
  tile[r][c + 1] = (__bf16)v.y;
  tile[r][c + 2] = (__bf16)v.z;
  tile[r][c + 3] = (__bf16)v.w;
  __syncthreads();

  bfloat* dst = WT + (size_t)(i0 + r) * BOTTLENECK + h0 + c;
  dst[0] = tile[c + 0][r];
  dst[1] = tile[c + 1][r];
  dst[2] = tile[c + 2][r];
  dst[3] = tile[c + 3][r];
}

// ---------------------------------------------------------------------------
// Kernel 2: z1 = x @ W^T + b_enc  (bf16 WMMA, fp32 accumulate)
// Wave tile: 16(M) x 64(N). Block: 8 waves -> 64(M) x 128(N).
// ---------------------------------------------------------------------------
__global__ __launch_bounds__(256) void k_encode(const float* __restrict__ x,
                                                const float* __restrict__ W,
                                                const float* __restrict__ b_enc,
                                                float* __restrict__ z1) {
  const int lane  = threadIdx.x & 31;
  const int wave  = threadIdx.x >> 5;
  const int waveM = wave >> 1;   // 0..3
  const int waveN = wave & 1;    // 0..1
  const int half  = lane >> 4;
  const int l16   = lane & 15;
  const int m0 = blockIdx.x * 64 + waveM * 16;    // batch rows
  const int n0 = blockIdx.y * 128 + waveN * 64;   // hidden cols

  const float* xrow = x + (size_t)(m0 + l16) * INPUT_DIM;
  const float* wr0  = W + (size_t)(n0 +  0 + l16) * INPUT_DIM;
  const float* wr1  = W + (size_t)(n0 + 16 + l16) * INPUT_DIM;
  const float* wr2  = W + (size_t)(n0 + 32 + l16) * INPUT_DIM;
  const float* wr3  = W + (size_t)(n0 + 48 + l16) * INPUT_DIM;

  v8f acc0 = {}, acc1 = {}, acc2 = {}, acc3 = {};
  for (int k = 0; k < INPUT_DIM; k += 32) {
    v16bf a  = frag_from_f32(xrow, k, half, 0.0f, false);
    v16bf b0 = frag_from_f32(wr0,  k, half, 0.0f, false);
    v16bf b1 = frag_from_f32(wr1,  k, half, 0.0f, false);
    v16bf b2 = frag_from_f32(wr2,  k, half, 0.0f, false);
    v16bf b3 = frag_from_f32(wr3,  k, half, 0.0f, false);
    acc0 = __builtin_amdgcn_wmma_f32_16x16x32_bf16(false, a, false, b0, (short)0, acc0, false, false);
    acc1 = __builtin_amdgcn_wmma_f32_16x16x32_bf16(false, a, false, b1, (short)0, acc1, false, false);
    acc2 = __builtin_amdgcn_wmma_f32_16x16x32_bf16(false, a, false, b2, (short)0, acc2, false, false);
    acc3 = __builtin_amdgcn_wmma_f32_16x16x32_bf16(false, a, false, b3, (short)0, acc3, false, false);
  }

  // C/D layout: lane -> (M = half*8 + vgpr, N = lane&15)
  const int cm = m0 + half * 8;
  const int cn = n0 + l16;
  const float be0 = b_enc[cn];
  const float be1 = b_enc[cn + 16];
  const float be2 = b_enc[cn + 32];
  const float be3 = b_enc[cn + 48];
#pragma unroll
  for (int r = 0; r < 8; r++) {
    const size_t o = (size_t)(cm + r) * BOTTLENECK;
    z1[o + cn]      = acc0[r] + be0;
    z1[o + cn + 16] = acc1[r] + be1;
    z1[o + cn + 32] = acc2[r] + be2;
    z1[o + cn + 48] = acc3[r] + be3;
  }
}

// ---------------------------------------------------------------------------
// Kernel 3: per-row radix select of the K_COUNT-th largest value of z1[row].
// Monotone key: u = sign ? ~bits : bits|0x80000000 (bigger key = bigger float).
// 4 passes of 8 bits, LDS histogram.
// ---------------------------------------------------------------------------
__global__ __launch_bounds__(256) void k_topk(const float* __restrict__ z1,
                                              float* __restrict__ thresh) {
  __shared__ unsigned hist[256];
  __shared__ unsigned s_state[2];
  const int tid = threadIdx.x;
  const float* row = z1 + (size_t)blockIdx.x * BOTTLENECK;

  unsigned prefix = 0;
  unsigned k = K_COUNT;
  for (int shift = 24; shift >= 0; shift -= 8) {
    hist[tid] = 0;
    __syncthreads();
    const unsigned mask_hi = (shift == 24) ? 0u : (0xFFFFFFFFu << (shift + 8));
    for (int j = tid; j < BOTTLENECK; j += 256) {
      unsigned u = __float_as_uint(row[j]);
      u = (u & 0x80000000u) ? ~u : (u | 0x80000000u);
      if ((u & mask_hi) == prefix)
        atomicAdd(&hist[(u >> shift) & 255u], 1u);
    }
    __syncthreads();
    if (tid == 0) {
      unsigned cnt = 0;
      int b = 255;
      for (; b > 0; b--) {
        if (cnt + hist[b] >= k) break;
        cnt += hist[b];
      }
      s_state[0] = prefix | ((unsigned)b << shift);
      s_state[1] = k - cnt;
    }
    __syncthreads();
    prefix = s_state[0];
    k = s_state[1];
    __syncthreads();
  }

  if (tid == 0) {
    const unsigned bits = (prefix & 0x80000000u) ? (prefix ^ 0x80000000u) : ~prefix;
    thresh[blockIdx.x] = __uint_as_float(bits);
  }
}

// ---------------------------------------------------------------------------
// Kernel 4: out = (z1 masked by per-row threshold) @ W + b_dec
// B operand comes from the pre-transposed bf16 WT so K(=h) is contiguous.
// ---------------------------------------------------------------------------
__global__ __launch_bounds__(256) void k_decode(const float* __restrict__ z1,
                                                const bfloat* __restrict__ WT,
                                                const float* __restrict__ thresh,
                                                const float* __restrict__ b_dec,
                                                float* __restrict__ out) {
  const int lane  = threadIdx.x & 31;
  const int wave  = threadIdx.x >> 5;
  const int waveM = wave >> 1;
  const int waveN = wave & 1;
  const int half  = lane >> 4;
  const int l16   = lane & 15;
  const int m0 = blockIdx.x * 64 + waveM * 16;    // batch rows
  const int n0 = blockIdx.y * 128 + waveN * 64;   // input_dim cols

  const float* arow = z1 + (size_t)(m0 + l16) * BOTTLENECK;
  const float  th   = thresh[m0 + l16];
  const bfloat* wr0 = WT + (size_t)(n0 +  0 + l16) * BOTTLENECK;
  const bfloat* wr1 = WT + (size_t)(n0 + 16 + l16) * BOTTLENECK;
  const bfloat* wr2 = WT + (size_t)(n0 + 32 + l16) * BOTTLENECK;
  const bfloat* wr3 = WT + (size_t)(n0 + 48 + l16) * BOTTLENECK;

  v8f acc0 = {}, acc1 = {}, acc2 = {}, acc3 = {};
  for (int k = 0; k < BOTTLENECK; k += 32) {
    v16bf a  = frag_from_f32(arow, k, half, th, true);   // fused k-WTA mask
    v16bf b0 = frag_from_bf16(wr0, k, half);
    v16bf b1 = frag_from_bf16(wr1, k, half);
    v16bf b2 = frag_from_bf16(wr2, k, half);
    v16bf b3 = frag_from_bf16(wr3, k, half);
    acc0 = __builtin_amdgcn_wmma_f32_16x16x32_bf16(false, a, false, b0, (short)0, acc0, false, false);
    acc1 = __builtin_amdgcn_wmma_f32_16x16x32_bf16(false, a, false, b1, (short)0, acc1, false, false);
    acc2 = __builtin_amdgcn_wmma_f32_16x16x32_bf16(false, a, false, b2, (short)0, acc2, false, false);
    acc3 = __builtin_amdgcn_wmma_f32_16x16x32_bf16(false, a, false, b3, (short)0, acc3, false, false);
  }

  const int cm = m0 + half * 8;
  const int cn = n0 + l16;
  const float bd0 = b_dec[cn];
  const float bd1 = b_dec[cn + 16];
  const float bd2 = b_dec[cn + 32];
  const float bd3 = b_dec[cn + 48];
#pragma unroll
  for (int r = 0; r < 8; r++) {
    const size_t o = (size_t)(cm + r) * INPUT_DIM;
    out[o + cn]      = acc0[r] + bd0;
    out[o + cn + 16] = acc1[r] + bd1;
    out[o + cn + 32] = acc2[r] + bd2;
    out[o + cn + 48] = acc3[r] + bd3;
  }
}

// ---------------------------------------------------------------------------
// Launch
// ---------------------------------------------------------------------------
extern "C" void kernel_launch(void* const* d_in, const int* in_sizes, int n_in,
                              void* d_out, int out_size, void* d_ws, size_t ws_size,
                              hipStream_t stream) {
  const float* x     = (const float*)d_in[0];
  const float* W     = (const float*)d_in[1];
  const float* b_enc = (const float*)d_in[2];
  const float* b_dec = (const float*)d_in[3];
  (void)in_sizes; (void)n_in; (void)out_size; (void)ws_size;

  char* ws = (char*)d_ws;
  bfloat* WT = (bfloat*)ws;
  const size_t WT_bytes = (size_t)INPUT_DIM * BOTTLENECK * sizeof(bfloat);
  float* z1 = (float*)(ws + WT_bytes);
  const size_t z1_bytes = (size_t)BATCH * BOTTLENECK * sizeof(float);
  float* thresh = (float*)(ws + WT_bytes + z1_bytes);
  float* out = (float*)d_out;

  k_transpose<<<dim3(BOTTLENECK / 32, INPUT_DIM / 32), 256, 0, stream>>>(W, WT);
  k_encode<<<dim3(BATCH / 64, BOTTLENECK / 128), 256, 0, stream>>>(x, W, b_enc, z1);
  k_topk<<<dim3(BATCH), 256, 0, stream>>>(z1, thresh);
  k_decode<<<dim3(BATCH / 64, INPUT_DIM / 128), 256, 0, stream>>>(z1, WT, thresh, b_dec, out);
}
